// _NLBlock_77481210020376
// MI455X (gfx1250) — matrix operand order repaired and checked by reading, not compiled, not run on hardware
//
#include <hip/hip_runtime.h>
#include <stdint.h>

// ---------- types matching gfx1250 WMMA fragment layouts ----------
typedef __attribute__((ext_vector_type(16))) __bf16          v16bf;
typedef __attribute__((ext_vector_type(8)))  float           v8f;
typedef __attribute__((ext_vector_type(8)))  unsigned short  u16x8;

union FragBF { u16x8 h[2]; v16bf v; };

__device__ __forceinline__ unsigned short f32_to_bf16(float f) {
    union { float f; uint32_t u; } c; c.f = f;
    uint32_t u = c.u;
    uint32_t r = u + 0x7fffu + ((u >> 16) & 1u);   // round-to-nearest-even
    return (unsigned short)(r >> 16);
}

// ---------- fp32 -> bf16 conversion ----------
__global__ __launch_bounds__(256) void k_cvt_bf16(const float* __restrict__ src,
                                                  unsigned short* __restrict__ dst, int n) {
    int i = blockIdx.x * blockDim.x + threadIdx.x;
    if (i < n) dst[i] = f32_to_bf16(src[i]);
}

// ---------- register-blocked NT GEMM ----------
// out[m,n] = alpha * sum_k A[m,k]*B[n,k] (+bias)(+resid)
// A: [M,K] bf16 row-major (lda=K), B: [N,K] bf16 row-major (ldb=K).
// Block = 256 threads = 8 waves arranged 4(M) x 2(N).
// Each wave owns a 2x4 grid of 16x16 accumulators -> 32 rows x 64 cols.
// Block macro-tile: 128 x 128.  grid = (N/128, M/128, nbatch).
#define MT 2
#define NT 4
__global__ __launch_bounds__(256) void k_gemm_nt(
    const unsigned short* __restrict__ A, int lda, size_t batchA,
    const unsigned short* __restrict__ B, int ldb, size_t batchB,
    const float* __restrict__ bias, const float* __restrict__ resid,
    float* __restrict__ outF, unsigned short* __restrict__ outB,
    int M, int N, int K, float alpha, int tposeS)
{
    const int lane  = threadIdx.x & 31;
    const int wave  = threadIdx.x >> 5;
    const int mwave = wave & 3;          // 0..3
    const int nwave = wave >> 2;         // 0..1
    const int z     = blockIdx.z;
    const int m0    = blockIdx.y * 128 + mwave * (16 * MT);   // wave's first row
    const int n0    = blockIdx.x * 128 + nwave * (16 * NT);   // wave's first col
    if (m0 >= M || n0 >= N) return;      // uniform per wave (EXEC stays all-ones)

    const unsigned short* Ab = A + (size_t)z * batchA;
    const unsigned short* Bb = B + (size_t)z * batchB;

    // A 16x32 bf16 fragment: lane L holds row (tile_m0+L%16); lanes<16 get K[0:8],K[16:24],
    // lanes>=16 get K[8:16],K[24:32]  -> two contiguous 16B loads per lane.
    const int arow = lane & 15;
    const int aoff = (lane >> 4) << 3;         // 0 or 8
    // B 32x16 bf16 fragment: lane n holds row (tile_n0+n%16); lanes<16 get K[0:8],K[16:24],
    // lanes>=16 get K[8:16],K[24:32] (same as A).
    const int brow = lane & 15;
    const int boff = (lane >> 4) << 3;         // 0 or 8

    const unsigned short* pa0 = Ab + (size_t)(m0 +  0 + arow) * lda + aoff;
    const unsigned short* pa1 = Ab + (size_t)(m0 + 16 + arow) * lda + aoff;
    const unsigned short* pb0 = Bb + (size_t)(n0 +  0 + brow) * ldb + boff;
    const unsigned short* pb1 = Bb + (size_t)(n0 + 16 + brow) * ldb + boff;
    const unsigned short* pb2 = Bb + (size_t)(n0 + 32 + brow) * ldb + boff;
    const unsigned short* pb3 = Bb + (size_t)(n0 + 48 + brow) * ldb + boff;

    v8f acc[MT][NT];
#pragma unroll
    for (int i = 0; i < MT; ++i)
#pragma unroll
        for (int j = 0; j < NT; ++j) acc[i][j] = (v8f){};

#pragma unroll 2
    for (int k0 = 0; k0 < K; k0 += 32) {
        FragBF a[MT], b[NT];
        a[0].h[0] = *(const u16x8*)(pa0);  a[0].h[1] = *(const u16x8*)(pa0 + 16);
        a[1].h[0] = *(const u16x8*)(pa1);  a[1].h[1] = *(const u16x8*)(pa1 + 16);
        b[0].h[0] = *(const u16x8*)(pb0);  b[0].h[1] = *(const u16x8*)(pb0 + 16);
        b[1].h[0] = *(const u16x8*)(pb1);  b[1].h[1] = *(const u16x8*)(pb1 + 16);
        b[2].h[0] = *(const u16x8*)(pb2);  b[2].h[1] = *(const u16x8*)(pb2 + 16);
        b[3].h[0] = *(const u16x8*)(pb3);  b[3].h[1] = *(const u16x8*)(pb3 + 16);
#pragma unroll
        for (int i = 0; i < MT; ++i)
#pragma unroll
            for (int j = 0; j < NT; ++j)
                acc[i][j] = __builtin_amdgcn_wmma_f32_16x16x32_bf16(
                    false, a[i].v, false, b[j].v, (short)0, acc[i][j], false, false);
        pa0 += 32; pa1 += 32; pb0 += 32; pb1 += 32; pb2 += 32; pb3 += 32;
        __builtin_prefetch(pa0 + 32, 0, 1);    // global_prefetch_b8
        __builtin_prefetch(pb0 + 32, 0, 1);
        __builtin_prefetch(pb2 + 32, 0, 1);
    }

    // C/D f32 16x16 layout: lane(0-15)=col N, VGPR r = row r; lanes 16-31 = rows 8-15.
    const int col = lane & 15;
    const int rb  = (lane >> 4) << 3;
    const size_t outBase = (size_t)M * (size_t)N * (size_t)z;
#pragma unroll
    for (int j = 0; j < NT; ++j) {
        const int nn = n0 + j * 16 + col;
        const float bv = bias ? bias[nn] : 0.0f;
#pragma unroll
        for (int i = 0; i < MT; ++i) {
#pragma unroll
            for (int r = 0; r < 8; ++r) {
                const int m = m0 + i * 16 + rb + r;
                float v = acc[i][j][r] * alpha + bv;
                if (resid) v += resid[(size_t)m * N + nn];
                if (outF) {
                    outF[outBase + (size_t)m * N + nn] = v;
                } else if (tposeS > 0) {       // transpose-store per sub-batch of tposeS rows
                    const int b2 = m / tposeS, s = m % tposeS;
                    outB[(size_t)b2 * (size_t)N * tposeS + (size_t)nn * tposeS + s] = f32_to_bf16(v);
                } else {
                    outB[(size_t)m * N + nn] = f32_to_bf16(v);
                }
            }
        }
    }
}

// ---------- row softmax: scores fp32 [rows, 2048] -> attn bf16 ----------
__global__ __launch_bounds__(256) void k_softmax2048(const float* __restrict__ scores,
                                                     unsigned short* __restrict__ attn) {
    __shared__ float red[256];
    const int row = blockIdx.x, tid = threadIdx.x;
    const float* sr = scores + (size_t)row * 2048;
    float v[8]; float mx = -3.4e38f;
#pragma unroll
    for (int i = 0; i < 8; ++i) { v[i] = sr[tid + i * 256]; mx = fmaxf(mx, v[i]); }
    red[tid] = mx; __syncthreads();
    for (int s = 128; s > 0; s >>= 1) { if (tid < s) red[tid] = fmaxf(red[tid], red[tid + s]); __syncthreads(); }
    mx = red[0]; __syncthreads();
    float sum = 0.f;
#pragma unroll
    for (int i = 0; i < 8; ++i) { v[i] = __expf(v[i] - mx); sum += v[i]; }
    red[tid] = sum; __syncthreads();
    for (int s = 128; s > 0; s >>= 1) { if (tid < s) red[tid] += red[tid + s]; __syncthreads(); }
    const float inv = 1.0f / red[0];
    unsigned short* ar = attn + (size_t)row * 2048;
#pragma unroll
    for (int i = 0; i < 8; ++i) ar[tid + i * 256] = f32_to_bf16(v[i] * inv);
}

// ---------- LayerNorm (C=1024) + ReLU, fp32 in -> bf16 out ----------
__global__ __launch_bounds__(256) void k_ln_relu1024(const float* __restrict__ xin,
                                                     const float* __restrict__ g,
                                                     const float* __restrict__ bta,
                                                     unsigned short* __restrict__ hout) {
    __shared__ float rs_[256], rq_[256];
    const int row = blockIdx.x, tid = threadIdx.x;
    const float* xr = xin + (size_t)row * 1024;
    float v[4], s = 0.f, q = 0.f;
#pragma unroll
    for (int i = 0; i < 4; ++i) { v[i] = xr[tid + i * 256]; s += v[i]; q += v[i] * v[i]; }
    rs_[tid] = s; rq_[tid] = q; __syncthreads();
    for (int st = 128; st > 0; st >>= 1) {
        if (tid < st) { rs_[tid] += rs_[tid + st]; rq_[tid] += rq_[tid + st]; }
        __syncthreads();
    }
    const float mu  = rs_[0] * (1.0f / 1024.0f);
    const float var = rq_[0] * (1.0f / 1024.0f) - mu * mu;
    const float rsg = rsqrtf(var + 1e-5f);
    unsigned short* hr = hout + (size_t)row * 1024;
#pragma unroll
    for (int i = 0; i < 4; ++i) {
        const int c = tid + i * 256;
        const float y = (v[i] - mu) * rsg * g[c] + bta[c];
        hr[c] = f32_to_bf16(fmaxf(y, 0.0f));
    }
}

// ---------------------------------------------------------------------------
extern "C" void kernel_launch(void* const* d_in, const int* in_sizes, int n_in,
                              void* d_out, int out_size, void* d_ws, size_t ws_size,
                              hipStream_t stream) {
    (void)in_sizes; (void)n_in; (void)out_size; (void)ws_size;
    const int B = 8, T = 1024, S = 2048, C = 1024;
    const int MX = B * T;      // 8192
    const int ML = B * S;      // 16384

    const float* x      = (const float*)d_in[0];
    const float* lfb    = (const float*)d_in[1];
    const float* thW    = (const float*)d_in[2];
    const float* thB    = (const float*)d_in[3];
    const float* phW    = (const float*)d_in[4];
    const float* phB    = (const float*)d_in[5];
    const float* giW    = (const float*)d_in[6];
    const float* giB    = (const float*)d_in[7];
    const float* lnG    = (const float*)d_in[8];
    const float* lnB    = (const float*)d_in[9];
    const float* fcW    = (const float*)d_in[10];
    const float* fcB    = (const float*)d_in[11];
    float*       out    = (float*)d_out;

    char* w = (char*)d_ws;
    size_t o = 0;
    unsigned short* xb   = (unsigned short*)(w + o); o += (size_t)MX * C * 2;
    unsigned short* lb   = (unsigned short*)(w + o); o += (size_t)ML * C * 2;
    unsigned short* wth  = (unsigned short*)(w + o); o += (size_t)C * C * 2;
    unsigned short* wph  = (unsigned short*)(w + o); o += (size_t)C * C * 2;
    unsigned short* wgi  = (unsigned short*)(w + o); o += (size_t)C * C * 2;
    unsigned short* wfc  = (unsigned short*)(w + o); o += (size_t)C * C * 2;
    unsigned short* thb  = (unsigned short*)(w + o); o += (size_t)MX * C * 2;   // theta bf16
    unsigned short* phb  = (unsigned short*)(w + o); o += (size_t)ML * C * 2;   // phi bf16
    unsigned short* giT  = (unsigned short*)(w + o); o += (size_t)ML * C * 2;   // gi^T per batch [C,S]
    float*          sc   = (float*)(w + o);          o += (size_t)MX * S * 4;   // scores fp32
    unsigned short* attn = (unsigned short*)(w + o); o += (size_t)MX * S * 2;   // attn bf16
    float*          nl   = (float*)(w + o);          o += (size_t)MX * C * 4;   // attn@gi fp32
    unsigned short* hb   = (unsigned short*)(w + o); o += (size_t)MX * C * 2;   // relu(ln) bf16

    // 1) fp32 -> bf16 conversions
    {
        int n;
        n = MX * C;  k_cvt_bf16<<<(n + 255) / 256, 256, 0, stream>>>(x,   xb,  n);
        n = ML * C;  k_cvt_bf16<<<(n + 255) / 256, 256, 0, stream>>>(lfb, lb,  n);
        n = C * C;
        k_cvt_bf16<<<(n + 255) / 256, 256, 0, stream>>>(thW, wth, n);
        k_cvt_bf16<<<(n + 255) / 256, 256, 0, stream>>>(phW, wph, n);
        k_cvt_bf16<<<(n + 255) / 256, 256, 0, stream>>>(giW, wgi, n);
        k_cvt_bf16<<<(n + 255) / 256, 256, 0, stream>>>(fcW, wfc, n);
    }

    // 2) theta = x @ thW^T + thB   [8192 x 1024] bf16
    k_gemm_nt<<<dim3(C / 128, MX / 128, 1), 256, 0, stream>>>(
        xb, C, 0, wth, C, 0, thB, nullptr, nullptr, thb, MX, C, C, 1.0f, 0);
    // 3) phi = lfb @ phW^T + phB   [16384 x 1024] bf16
    k_gemm_nt<<<dim3(C / 128, ML / 128, 1), 256, 0, stream>>>(
        lb, C, 0, wph, C, 0, phB, nullptr, nullptr, phb, ML, C, C, 1.0f, 0);
    // 4) giT = (lfb @ giW^T + giB) transposed per batch -> [B][C][S] bf16
    k_gemm_nt<<<dim3(C / 128, ML / 128, 1), 256, 0, stream>>>(
        lb, C, 0, wgi, C, 0, giB, nullptr, nullptr, giT, ML, C, C, 1.0f, S);

    // 5) scores = sqrt(C) * theta @ phi^T  per batch  [B][T][S] fp32
    k_gemm_nt<<<dim3(S / 128, T / 128, B), 256, 0, stream>>>(
        thb, C, (size_t)T * C, phb, C, (size_t)S * C,
        nullptr, nullptr, sc, nullptr, T, S, C, 32.0f, 0);

    // 6) softmax rows -> bf16 attn
    k_softmax2048<<<MX, 256, 0, stream>>>(sc, attn);

    // 7) nl = attn @ giT^T per batch  [B][T][C] fp32   (NT: K = S contiguous in both)
    k_gemm_nt<<<dim3(C / 128, T / 128, B), 256, 0, stream>>>(
        attn, S, (size_t)T * S, giT, S, (size_t)C * S,
        nullptr, nullptr, nl, nullptr, T, C, S, 1.0f, 0);

    // 8) h = relu(layernorm(nl)) -> bf16
    k_ln_relu1024<<<MX, 256, 0, stream>>>(nl, lnG, lnB, hb);

    // 9) out = h @ fcW^T + fcB + x   [8192 x 1024] fp32
    k_gemm_nt<<<dim3(C / 128, MX / 128, 1), 256, 0, stream>>>(
        hb, C, 0, wfc, C, 0, fcB, x, out, nullptr, MX, C, C, 1.0f, 0);
}
